// TransformerDecoder_53360673685879
// MI455X (gfx1250) — matrix operand order repaired and checked by reading, not compiled, not run on hardware
//
#include <hip/hip_runtime.h>
#include <hip/hip_bf16.h>

// ---------------- CDNA5 WMMA types ----------------
typedef __attribute__((ext_vector_type(16))) __bf16 v16bf;
typedef __attribute__((ext_vector_type(8)))  float  v8f;
typedef __attribute__((ext_vector_type(4)))  unsigned int u32x4;
typedef __attribute__((ext_vector_type(4)))  float  f32x4;

union FragBF {
    __bf16 h[16];
    v16bf  v;
    u32x4  q[2];
};
union PackBF2 {
    __bf16 h[2];
    unsigned int u;
};

__device__ inline v8f wmma_bf16(v16bf a, v16bf b, v8f c) {
    // (neg_a, A, neg_b, B, c_mod, C, reuse_a, reuse_b)
    return __builtin_amdgcn_wmma_f32_16x16x32_bf16(false, a, false, b, (short)0, c, false, false);
}

// Build a v16bf fragment from two runs of 8 consecutive floats (vectorized b128 loads).
__device__ inline v16bf cvt16(const float* g0, const float* g1) {
    f32x4 a = *(const f32x4*)g0;
    f32x4 b = *(const f32x4*)(g0 + 4);
    f32x4 c = *(const f32x4*)g1;
    f32x4 d = *(const f32x4*)(g1 + 4);
    FragBF f;
    f.h[0]=(__bf16)a.x;  f.h[1]=(__bf16)a.y;  f.h[2]=(__bf16)a.z;  f.h[3]=(__bf16)a.w;
    f.h[4]=(__bf16)b.x;  f.h[5]=(__bf16)b.y;  f.h[6]=(__bf16)b.z;  f.h[7]=(__bf16)b.w;
    f.h[8]=(__bf16)c.x;  f.h[9]=(__bf16)c.y;  f.h[10]=(__bf16)c.z; f.h[11]=(__bf16)c.w;
    f.h[12]=(__bf16)d.x; f.h[13]=(__bf16)d.y; f.h[14]=(__bf16)d.z; f.h[15]=(__bf16)d.w;
    return f.v;
}
// A-matrix 16x32 lane layout: K = h + 16*(lane>>4) (16 contiguous K), kb=16*(lane>>4)
__device__ inline v16bf make_fragA_f32(const float* row, int lane) {
    const int kb = 16 * (lane >> 4);
    return cvt16(row + kb, row + kb + 8);
}
// B-matrix 32x16 lane layout: lane%16 = column N, K = h + 16*(lane>>4) (16 contiguous K)
__device__ inline v16bf make_fragB_f32(const float* col, int lane) {
    const int kb = 16 * (lane >> 4);
    return cvt16(col + kb, col + kb + 8);
}

// ---------------- problem constants ----------------
#define QN   300
#define BN   16
#define SN   1024
#define DM   256
#define D2M  512
#define HN   8
#define HD   64
#define FFNN 1024
#define NL   6
#define MQB  (QN * BN)     // 4800
#define MSB  (SN * BN)     // 16384

// =====================================================================
// Generic tiled WMMA GEMM: C[M,N] = act(A[:, aoff:aoff+K] @ W[K,N] + bias)
// A row stride = lda. M%64==0, N%128==0, K%32==0 (guaranteed by caller).
// Block: 256 threads = 8 waves, each computing a 32x32 tile (2x2 WMMA with
// operand reuse); block tile 64x128. Double-buffered LDS (24KB) so staging
// of tile i+1 (b128 global loads + packed bf16 converts) overlaps the WMMA
// chain of tile i; one barrier per K-step.
// =====================================================================
__global__ __launch_bounds__(256)
void wmma_gemm_kernel(const float* __restrict__ A, const float* __restrict__ W,
                      const float* __restrict__ bias, float* __restrict__ C,
                      int M, int N, int K, int lda, int aoff, int relu)
{
    __shared__ __align__(16) __bf16 sA[2][64][32];
    __shared__ __align__(16) __bf16 sB[2][128][32];   // transposed: [n][k]

    const int tid  = threadIdx.x;
    const int lane = tid & 31;
    const int wid  = tid >> 5;
    const int m0 = blockIdx.y * 64;
    const int n0 = blockIdx.x * 128;
    const int wm = (wid & 1) * 32;     // wave row offset (2 rows of wave tiles)
    const int wn = (wid >> 1) * 32;    // wave col offset (4 cols of wave tiles)

    // fixed per-thread staging assignments (no exec-masked loops)
    const int ar  = tid >> 2;           // A: row 0..63
    const int ac  = (tid & 3) << 3;     // A: col 0,8,16,24
    const int bkp = (tid >> 4) << 1;    // B: k pair base 0,2,...,30
    const int bnn = (tid & 15) << 3;    // B: n 0,8,...,120
    const float* gA = A + (size_t)(m0 + ar) * lda + aoff + ac;
    const float* gW = W + (size_t)bkp * N + n0 + bnn;

    auto stage = [&](int buf, int kt) {
        // ---- A tile 64x32: 2x b128 loads -> 8 bf16 -> 1x ds_store_b128 ----
        {
            f32x4 a = *(const f32x4*)(gA + kt);
            f32x4 b = *(const f32x4*)(gA + kt + 4);
            FragBF t;
            t.h[0]=(__bf16)a.x; t.h[1]=(__bf16)a.y; t.h[2]=(__bf16)a.z; t.h[3]=(__bf16)a.w;
            t.h[4]=(__bf16)b.x; t.h[5]=(__bf16)b.y; t.h[6]=(__bf16)b.z; t.h[7]=(__bf16)b.w;
            *(u32x4*)&sA[buf][ar][ac] = t.q[0];
        }
        // ---- W tile 32x128 transposed: 4x b128 loads, 8x packed b32 stores ----
        {
            const float* gw = gW + (size_t)kt * N;   // rows kt+bkp, kt+bkp+1
            f32x4 r0a = *(const f32x4*)gw;
            f32x4 r0b = *(const f32x4*)(gw + 4);
            f32x4 r1a = *(const f32x4*)(gw + N);
            f32x4 r1b = *(const f32x4*)(gw + N + 4);
            PackBF2 p;
            p.h[0]=(__bf16)r0a.x; p.h[1]=(__bf16)r1a.x; *(unsigned int*)&sB[buf][bnn+0][bkp] = p.u;
            p.h[0]=(__bf16)r0a.y; p.h[1]=(__bf16)r1a.y; *(unsigned int*)&sB[buf][bnn+1][bkp] = p.u;
            p.h[0]=(__bf16)r0a.z; p.h[1]=(__bf16)r1a.z; *(unsigned int*)&sB[buf][bnn+2][bkp] = p.u;
            p.h[0]=(__bf16)r0a.w; p.h[1]=(__bf16)r1a.w; *(unsigned int*)&sB[buf][bnn+3][bkp] = p.u;
            p.h[0]=(__bf16)r0b.x; p.h[1]=(__bf16)r1b.x; *(unsigned int*)&sB[buf][bnn+4][bkp] = p.u;
            p.h[0]=(__bf16)r0b.y; p.h[1]=(__bf16)r1b.y; *(unsigned int*)&sB[buf][bnn+5][bkp] = p.u;
            p.h[0]=(__bf16)r0b.z; p.h[1]=(__bf16)r1b.z; *(unsigned int*)&sB[buf][bnn+6][bkp] = p.u;
            p.h[0]=(__bf16)r0b.w; p.h[1]=(__bf16)r1b.w; *(unsigned int*)&sB[buf][bnn+7][bkp] = p.u;
            if (kt + 32 < K) {               // CDNA5 prefetch of tiles two steps ahead
                __builtin_prefetch(gA + kt + 32, 0, 1);
                __builtin_prefetch(gw + (size_t)32 * N, 0, 1);
            }
        }
    };

    v8f acc[2][2] = {};

    stage(0, 0);
    __syncthreads();

    for (int k0 = 0; k0 < K; k0 += 32) {
        const int buf = (k0 >> 5) & 1;
        if (k0 + 32 < K) stage(buf ^ 1, k0 + 32);   // overlap next-tile staging

        const int ka = 16 * (lane >> 4);
        const int kb = 16 * (lane >> 4);
        const __bf16* pa0 = &sA[buf][wm + (lane & 15)][0];
        const __bf16* pa1 = &sA[buf][wm + 16 + (lane & 15)][0];
        const __bf16* pb0 = &sB[buf][wn + (lane & 15)][0];
        const __bf16* pb1 = &sB[buf][wn + 16 + (lane & 15)][0];
        FragBF a0, a1, b0, b1;
        a0.q[0] = *(const u32x4*)(pa0 + ka); a0.q[1] = *(const u32x4*)(pa0 + ka + 8);
        a1.q[0] = *(const u32x4*)(pa1 + ka); a1.q[1] = *(const u32x4*)(pa1 + ka + 8);
        b0.q[0] = *(const u32x4*)(pb0 + kb); b0.q[1] = *(const u32x4*)(pb0 + kb + 8);
        b1.q[0] = *(const u32x4*)(pb1 + kb); b1.q[1] = *(const u32x4*)(pb1 + kb + 8);
        acc[0][0] = wmma_bf16(a0.v, b0.v, acc[0][0]);
        acc[0][1] = wmma_bf16(a0.v, b1.v, acc[0][1]);
        acc[1][0] = wmma_bf16(a1.v, b0.v, acc[1][0]);
        acc[1][1] = wmma_bf16(a1.v, b1.v, acc[1][1]);

        __syncthreads();   // next-tile stores visible; this buf free after next step
    }

    #pragma unroll
    for (int fi = 0; fi < 2; ++fi)
        #pragma unroll
        for (int fj = 0; fj < 2; ++fj) {
            const int col = n0 + wn + fj * 16 + (lane & 15);
            const float bv = bias ? bias[col] : 0.0f;
            #pragma unroll
            for (int i = 0; i < 8; ++i) {
                const int row = m0 + wm + fi * 16 + i + 8 * (lane >> 4);
                float v = acc[fi][fj][i] + bv;
                if (relu) v = fmaxf(v, 0.0f);
                C[(size_t)row * N + col] = v;
            }
        }
}

// =====================================================================
// Fused attention: one wave per (b, h, 16-query tile). Two-pass with the
// full logit row (16 x Sk, Sk<=1024) resident in LDS (64KB of the 320KB WGP LDS).
//   logits = (Q @ K^T) * scale (+ bias[b,q,k]) ; softmax ; O = P @ V
// Layout of Q/K/V: [L*B, 512], row = l*B + b, head h at cols h*64..h*64+63.
// =====================================================================
__global__ __launch_bounds__(32)
void attn_kernel(const float* __restrict__ Qp, const float* __restrict__ Kp,
                 const float* __restrict__ Vp, const float* __restrict__ bias,
                 float* __restrict__ Out, int Lq, int Sk, int Bb, int H)
{
    __shared__ __align__(16) float sL[16][1024];

    const int lane = threadIdx.x;
    const int qt   = blockIdx.x;
    const int b    = blockIdx.y / H;
    const int h    = blockIdx.y % H;
    const int D2v  = H * HD;
    const int nKt  = (Sk + 15) >> 4;
    const int Sp   = ((Sk + 31) >> 5) << 5;   // pad to 32 for the P@V pass
    const float scale = rsqrtf((float)HD);

    // ---- Q fragments (A-role), 16x64 split into two 16x32 ----
    v16bf qa[2];
    {
        int q = qt * 16 + (lane & 15); if (q >= Lq) q = Lq - 1;
        const float* base = Qp + (size_t)(q * Bb + b) * D2v + h * HD;
        qa[0] = make_fragA_f32(base, lane);
        qa[1] = make_fragA_f32(base + 32, lane);
    }

    // ---- pass 1: logits into LDS ----
    for (int kt = 0; kt < nKt; ++kt) {
        v8f acc = {};
        int s = kt * 16 + (lane & 15); if (s >= Sk) s = Sk - 1;
        const float* kbase = Kp + (size_t)(s * Bb + b) * D2v + h * HD;
        acc = wmma_bf16(qa[0], make_fragB_f32(kbase, lane), acc);
        acc = wmma_bf16(qa[1], make_fragB_f32(kbase + 32, lane), acc);

        const int col = kt * 16 + (lane & 15);
        #pragma unroll
        for (int i = 0; i < 8; ++i) {
            const int row = i + 8 * (lane >> 4);
            float v = acc[i] * scale;
            if (bias) {
                const int q = qt * 16 + row;
                if (q < Lq && col < Sk)
                    v += bias[((size_t)b * Lq + q) * Sk + col];
            }
            sL[row][col] = v;
        }
    }
    // mask padded key columns
    for (int kk = Sk + lane; kk < Sp; kk += 32)
        #pragma unroll
        for (int r = 0; r < 16; ++r) sL[r][kk] = -1e30f;
    asm volatile("s_wait_dscnt 0" ::: "memory");   // LDS writes visible (same-wave, in order)

    // ---- softmax over each of the 16 rows (wave32 shuffle reductions) ----
    for (int r = 0; r < 16; ++r) {
        float m = -1e30f;
        for (int kk = lane; kk < Sp; kk += 32) m = fmaxf(m, sL[r][kk]);
        #pragma unroll
        for (int off = 16; off; off >>= 1) m = fmaxf(m, __shfl_xor(m, off, 32));
        float sum = 0.0f;
        for (int kk = lane; kk < Sp; kk += 32) {
            float e = __expf(sL[r][kk] - m);
            sL[r][kk] = e;
            sum += e;
        }
        #pragma unroll
        for (int off = 16; off; off >>= 1) sum += __shfl_xor(sum, off, 32);
        const float inv = 1.0f / sum;
        for (int kk = lane; kk < Sp; kk += 32) sL[r][kk] *= inv;
    }

    // ---- pass 2: O = P @ V (16 x 64 output = 4 WMMA tiles) ----
    v8f oacc[4] = {};
    for (int c = 0; c < Sp; c += 32) {
        const v16bf pf = make_fragA_f32(&sL[lane & 15][c], lane);   // ds_load_b128 x4
        const int kb = 16 * (lane >> 4);
        const bool full = (c + 32 <= Sk);
        #pragma unroll
        for (int t = 0; t < 4; ++t) {
            FragBF vf;
            const int n = t * 16 + (lane & 15);
            if (full) {   // uniform branch: strided gather, no clamping
                const float* vb = Vp + ((size_t)((c + kb) * Bb + b)) * D2v + h * HD + n;
                const size_t stride = (size_t)Bb * D2v;
                #pragma unroll
                for (int i = 0; i < 16; ++i)
                    vf.h[i] = (__bf16)vb[(size_t)i * stride];
            } else {
                #pragma unroll
                for (int i = 0; i < 16; ++i) {
                    int s = c + kb + i; if (s >= Sk) s = Sk - 1;
                    vf.h[i] = (__bf16)Vp[(size_t)(s * Bb + b) * D2v + h * HD + n];
                }
            }
            oacc[t] = wmma_bf16(pf, vf.v, oacc[t]);
        }
    }

    #pragma unroll
    for (int t = 0; t < 4; ++t)
        #pragma unroll
        for (int i = 0; i < 8; ++i) {
            const int row = i + 8 * (lane >> 4);
            const int q = qt * 16 + row;
            if (q < Lq)
                Out[(size_t)(q * Bb + b) * D2v + h * HD + t * 16 + (lane & 15)] = oacc[t][i];
        }
}

// ========================= elementwise / small kernels =========================
__global__ void ew_add_kernel(const float* a, const float* b, float* c, long n) {
    long i = (long)blockIdx.x * blockDim.x + threadIdx.x;
    if (i < n) c[i] = a[i] + b[i];
}

// dst = out + concat(query_pos, query_pos)
__global__ void ew_qsa_kernel(const float* o, const float* qp, float* dst, long n) {
    long i = (long)blockIdx.x * blockDim.x + threadIdx.x;
    if (i >= n) return;
    long m = i >> 9; int c = (int)(i & 511);
    dst[i] = o[i] + qp[m * DM + (c & 255)];
}

// dst = out + qp2 + qs2
__global__ void ew_qin_kernel(const float* o, const float* qp, const float* qs,
                              float* dst, long n) {
    long i = (long)blockIdx.x * blockDim.x + threadIdx.x;
    if (i >= n) return;
    long m = i >> 9; int c = (int)(i & 511);
    dst[i] = o[i] + qp[m * DM + (c & 255)] + qs[m * DM + (c & 255)];
}

// qs = sine_embed(sigmoid(ref_before)) * pos_transform   (d=256, fd=128)
__global__ void qs_kernel(const float* refpt, const float* pt, float* qs, long n) {
    long i = (long)blockIdx.x * blockDim.x + threadIdx.x;
    if (i >= n) return;
    long m = i >> 8; int c = (int)(i & 255);
    float p; int cc;
    if (c < 128) { p = refpt[m * 2 + 1]; cc = c; }        // y-half first
    else         { p = refpt[m * 2 + 0]; cc = c - 128; }  // then x-half
    const float dimt = __powf(10000.0f, (float)(cc & ~1) * (1.0f / 128.0f));
    const float ang  = p * 6.283185307179586f / dimt;
    const float se   = (cc & 1) ? __cosf(ang) : __sinf(ang);
    qs[i] = se * pt[i];
}

// ref_point head: ref_before / sigmoid  (hidden[4800,256] @ W2[256,2] + b2)
__global__ void rp_head_kernel(const float* h, const float* W2, const float* b2,
                               float* refb, float* refpt, int M) {
    int m = blockIdx.x * blockDim.x + threadIdx.x;
    if (m >= M) return;
    for (int j = 0; j < 2; ++j) {
        float s = b2[j];
        for (int k = 0; k < DM; ++k) s += h[(size_t)m * DM + k] * W2[k * 2 + j];
        refb[m * 2 + j]  = s;
        refpt[m * 2 + j] = 1.0f / (1.0f + __expf(-s));
    }
}

// bbox head -> boxes[B,Q,4] (xyxy), cs[B,Q] = cx+cy
__global__ void bb_head_kernel(const float* h, const float* W2, const float* b2,
                               const float* refb, float* boxes, float* cs, int M) {
    int m = blockIdx.x * blockDim.x + threadIdx.x;
    if (m >= M) return;
    float t[4];
    for (int j = 0; j < 4; ++j) {
        float s = b2[j];
        for (int k = 0; k < DM; ++k) s += h[(size_t)m * DM + k] * W2[k * 4 + j];
        t[j] = s;
    }
    t[0] += refb[m * 2 + 0];
    t[1] += refb[m * 2 + 1];
    const float cx = 1.0f / (1.0f + __expf(-t[0]));
    const float cy = 1.0f / (1.0f + __expf(-t[1]));
    const float w  = 1.0f / (1.0f + __expf(-t[2]));
    const float hh = 1.0f / (1.0f + __expf(-t[3]));
    const int q = m / BN, b = m % BN;
    float* bx = boxes + ((size_t)b * QN + q) * 4;
    bx[0] = cx - 0.5f * w;  bx[1] = cy - 0.5f * hh;
    bx[2] = cx + 0.5f * w;  bx[3] = cy + 0.5f * hh;
    cs[(size_t)b * QN + q] = cx + cy;
}

// bias[b,i,j] = IoU(box_i, box_j) + (cs_i > cs_j)
__global__ void bias_kernel(const float* boxes, const float* cs, float* bias, long n) {
    long idx = (long)blockIdx.x * blockDim.x + threadIdx.x;
    if (idx >= n) return;
    int j = (int)(idx % QN); long r = idx / QN;
    int i = (int)(r % QN);  int b = (int)(r / QN);
    const float* bi = boxes + ((size_t)b * QN + i) * 4;
    const float* bj = boxes + ((size_t)b * QN + j) * 4;
    const float ai = (bi[2] - bi[0]) * (bi[3] - bi[1]);
    const float aj = (bj[2] - bj[0]) * (bj[3] - bj[1]);
    const float w  = fmaxf(fminf(bi[2], bj[2]) - fmaxf(bi[0], bj[0]), 0.0f);
    const float hh = fmaxf(fminf(bi[3], bj[3]) - fmaxf(bi[1], bj[1]), 0.0f);
    const float inter = w * hh;
    const float iou = inter / (ai + aj - inter);
    const float ind = (cs[(size_t)b * QN + i] > cs[(size_t)b * QN + j]) ? 1.0f : 0.0f;
    bias[idx] = iou + ind;
}

// out[row] = LayerNorm(out[row] + add[row]), width 512; one wave per row
__global__ __launch_bounds__(128)
void ln_res_kernel(float* out, const float* add, int M) {
    const int lane = threadIdx.x & 31;
    const int row  = blockIdx.x * 4 + (threadIdx.x >> 5);
    if (row >= M) return;
    float* o = out + (size_t)row * D2M;
    const float* a = add + (size_t)row * D2M;
    float vals[16]; float s = 0.0f;
    #pragma unroll
    for (int i = 0; i < 16; ++i) {
        float v = o[lane + 32 * i] + a[lane + 32 * i];
        vals[i] = v; s += v;
    }
    #pragma unroll
    for (int off = 16; off; off >>= 1) s += __shfl_xor(s, off, 32);
    const float mean = s * (1.0f / D2M);
    float vs = 0.0f;
    #pragma unroll
    for (int i = 0; i < 16; ++i) { float d = vals[i] - mean; vs += d * d; }
    #pragma unroll
    for (int off = 16; off; off >>= 1) vs += __shfl_xor(vs, off, 32);
    const float inv = rsqrtf(vs * (1.0f / D2M) + 1e-5f);
    #pragma unroll
    for (int i = 0; i < 16; ++i) o[lane + 32 * i] = (vals[i] - mean) * inv;
}

// ============================ host orchestration ============================
extern "C" void kernel_launch(void* const* d_in, const int* in_sizes, int n_in,
                              void* d_out, int out_size, void* d_ws, size_t ws_size,
                              hipStream_t stream) {
    (void)in_sizes; (void)n_in; (void)out_size; (void)ws_size;

    const float* tgt       = (const float*)d_in[0];
    const float* memory    = (const float*)d_in[1];
    const float* pos       = (const float*)d_in[2];
    const float* query_pos = (const float*)d_in[3];
    const float* qs_W1 = (const float*)d_in[4];  const float* qs_b1 = (const float*)d_in[5];
    const float* qs_W2 = (const float*)d_in[6];  const float* qs_b2 = (const float*)d_in[7];
    const float* rp_W1 = (const float*)d_in[8];  const float* rp_b1 = (const float*)d_in[9];
    const float* rp_W2 = (const float*)d_in[10]; const float* rp_b2 = (const float*)d_in[11];
    const float* bb_W1 = (const float*)d_in[12]; const float* bb_b1 = (const float*)d_in[13];
    const float* bb_W2 = (const float*)d_in[14]; const float* bb_b2 = (const float*)d_in[15];
    const float* sa_Wq = (const float*)d_in[16]; const float* sa_Wk = (const float*)d_in[17];
    const float* sa_Wv = (const float*)d_in[18]; const float* sa_Wo = (const float*)d_in[19];
    const float* ca_Wq = (const float*)d_in[20]; const float* ca_Wk = (const float*)d_in[21];
    const float* ca_Wv = (const float*)d_in[22]; const float* ca_Wo = (const float*)d_in[23];
    const float* ff_W1 = (const float*)d_in[24]; const float* ff_b1 = (const float*)d_in[25];
    const float* ff_W2 = (const float*)d_in[26]; const float* ff_b2 = (const float*)d_in[27];

    float* out = (float*)d_out;   // running decoder state [Q,B,512], updated in place

    // ---- workspace carve-out ----
    float* wsp = (float*)d_ws; size_t cur = 0;
    auto alloc = [&](size_t nf) { float* p = wsp + cur; cur += nf; return p; };
    float* mp    = alloc((size_t)MSB * D2M);   // memory + pos
    float* h1    = alloc((size_t)MQB * DM);    // shared MLP hidden
    float* pt    = alloc((size_t)MQB * DM);    // pos_transform
    float* qsv   = alloc((size_t)MQB * DM);    // qs
    float* refb  = alloc((size_t)MQB * 2);
    float* refpt = alloc((size_t)MQB * 2);
    float* boxes = alloc((size_t)BN * QN * 4);
    float* csb   = alloc((size_t)BN * QN);
    float* biasb = alloc((size_t)BN * QN * QN);
    float* qbuf  = alloc((size_t)MQB * D2M);   // attn q-input, then attn output
    float* Pq    = alloc((size_t)MQB * D2M);   // Q projection, then out-proj result
    float* Pk    = alloc((size_t)MSB * D2M);
    float* Pv    = alloc((size_t)MSB * D2M);
    float* ffh   = alloc((size_t)MQB * FFNN);

    auto gemm = [&](const float* A, const float* W, const float* bias, float* C,
                    int M, int N, int K, int lda, int aoff, int relu) {
        wmma_gemm_kernel<<<dim3(N / 128, M / 64), 256, 0, stream>>>(
            A, W, bias, C, M, N, K, lda, aoff, relu);
    };
    auto ew = [&](long n) { return (unsigned)((n + 255) / 256); };

    // output = tgt
    hipMemcpyAsync(out, tgt, (size_t)MQB * D2M * sizeof(float),
                   hipMemcpyDeviceToDevice, stream);
    // mp = memory + pos
    { long n = (long)MSB * D2M; ew_add_kernel<<<ew(n), 256, 0, stream>>>(memory, pos, mp, n); }
    // ref_point head (constant across layers)
    gemm(query_pos, rp_W1, rp_b1, h1, MQB, DM, DM, DM, 0, 1);
    rp_head_kernel<<<(MQB + 255) / 256, 256, 0, stream>>>(h1, rp_W2, rp_b2, refb, refpt, MQB);

    const int nQt = (QN + 15) / 16;   // 19 query tiles
    for (int l = 0; l < NL; ++l) {
        const float* saWq = sa_Wq + (size_t)l * D2M * D2M;
        const float* saWk = sa_Wk + (size_t)l * D2M * D2M;
        const float* saWv = sa_Wv + (size_t)l * D2M * D2M;
        const float* saWo = sa_Wo + (size_t)l * D2M * D2M;
        const float* caWq = ca_Wq + (size_t)l * D2M * D2M;
        const float* caWk = ca_Wk + (size_t)l * D2M * D2M;
        const float* caWv = ca_Wv + (size_t)l * D2M * D2M;
        const float* caWo = ca_Wo + (size_t)l * D2M * D2M;
        const float* fW1  = ff_W1 + (size_t)l * D2M * FFNN;
        const float* fb1  = ff_b1 + (size_t)l * FFNN;
        const float* fW2  = ff_W2 + (size_t)l * FFNN * D2M;
        const float* fb2  = ff_b2 + (size_t)l * D2M;

        // ---- query-scale MLP + sine modulation ----
        gemm(out, qs_W1, qs_b1, h1, MQB, DM, DM, D2M, DM, 1);   // uses output[:,256:]
        gemm(h1, qs_W2, qs_b2, pt, MQB, DM, DM, DM, 0, 0);
        { long n = (long)MQB * DM; qs_kernel<<<ew(n), 256, 0, stream>>>(refpt, pt, qsv, n); }

        // ---- bbox head -> iou/cxcy attention bias ----
        gemm(out, bb_W1, bb_b1, h1, MQB, DM, DM, D2M, DM, 1);
        bb_head_kernel<<<(MQB + 255) / 256, 256, 0, stream>>>(h1, bb_W2, bb_b2, refb,
                                                              boxes, csb, MQB);
        { long n = (long)BN * QN * QN; bias_kernel<<<ew(n), 256, 0, stream>>>(boxes, csb, biasb, n); }

        // ---- self-attention ----
        { long n = (long)MQB * D2M; ew_qsa_kernel<<<ew(n), 256, 0, stream>>>(out, query_pos, qbuf, n); }
        gemm(qbuf, saWq, nullptr, Pq, MQB, D2M, D2M, D2M, 0, 0);
        gemm(qbuf, saWk, nullptr, Pk, MQB, D2M, D2M, D2M, 0, 0);
        gemm(out,  saWv, nullptr, Pv, MQB, D2M, D2M, D2M, 0, 0);
        attn_kernel<<<dim3(nQt, BN * HN), 32, 0, stream>>>(Pq, Pk, Pv, biasb, qbuf,
                                                           QN, QN, BN, HN);
        gemm(qbuf, saWo, nullptr, Pq, MQB, D2M, D2M, D2M, 0, 0);
        ln_res_kernel<<<(MQB + 3) / 4, 128, 0, stream>>>(out, Pq, MQB);

        // ---- cross-attention (sine-modulated query) ----
        { long n = (long)MQB * D2M; ew_qin_kernel<<<ew(n), 256, 0, stream>>>(out, query_pos, qsv, qbuf, n); }
        gemm(qbuf,   caWq, nullptr, Pq, MQB, D2M, D2M, D2M, 0, 0);
        gemm(mp,     caWk, nullptr, Pk, MSB, D2M, D2M, D2M, 0, 0);
        gemm(memory, caWv, nullptr, Pv, MSB, D2M, D2M, D2M, 0, 0);
        attn_kernel<<<dim3(nQt, BN * HN), 32, 0, stream>>>(Pq, Pk, Pv, nullptr, qbuf,
                                                           QN, SN, BN, HN);
        gemm(qbuf, caWo, nullptr, Pq, MQB, D2M, D2M, D2M, 0, 0);
        ln_res_kernel<<<(MQB + 3) / 4, 128, 0, stream>>>(out, Pq, MQB);

        // ---- FFN ----
        gemm(out, fW1, fb1, ffh, MQB, FFNN, D2M, D2M, 0, 1);
        gemm(ffh, fW2, fb2, Pq,  MQB, D2M, FFNN, FFNN, 0, 0);
        ln_res_kernel<<<(MQB + 3) / 4, 128, 0, stream>>>(out, Pq, MQB);
    }
}
